// CAM_11364483465724
// MI455X (gfx1250) — matrix operand order repaired and checked by reading, not compiled, not run on hardware
//
#include <hip/hip_runtime.h>
#include <hip/hip_bf16.h>
#include <math.h>

// CDNA5 / gfx1250, wave32. CAM (channel attention) with bf16x3-split WMMA:
// fp32 operands are split hi/lo into bf16 planes; each GEMM is computed as
// Ahi*Bhi + Ahi*Blo + Alo*Bhi accumulated in f32 via v_wmma_f32_16x16x32_bf16.

typedef __attribute__((ext_vector_type(8)))  float   v8f;
typedef __attribute__((ext_vector_type(4)))  __bf16  v4bf;
typedef __attribute__((ext_vector_type(8)))  __bf16  v8bf;
typedef __attribute__((ext_vector_type(16))) __bf16  v16bf;

#define B_   32
#define N_   4096   // H*W
#define C_   256
#define LDP  56     // LDS row pitch in bf16 (112 B): 16B-aligned, 28u%64 injective

static __device__ __forceinline__ v16bf ld16(const __bf16* p) {
    // K-contiguous 16 bf16 as two aligned 16B loads (A operand: K..K+7, K+16..K+23
    // callers pass the right pair; B operand: same K..K+7, K+16..K+23 pair).
    v8bf a = *(const v8bf*)p;
    v8bf b = *(const v8bf*)(p + 8);
    return __builtin_shufflevector(a, b, 0,1,2,3,4,5,6,7,8,9,10,11,12,13,14,15);
}
static __device__ __forceinline__ v16bf ld16_split(const __bf16* p0, const __bf16* p1) {
    v8bf a = *(const v8bf*)p0;
    v8bf b = *(const v8bf*)p1;
    return __builtin_shufflevector(a, b, 0,1,2,3,4,5,6,7,8,9,10,11,12,13,14,15);
}
static __device__ __forceinline__ v8f wmma_bf(v16bf a, v16bf b, v8f c) {
    return __builtin_amdgcn_wmma_f32_16x16x32_bf16(
        false, a, false, b, (short)0, c, false, false);
}
static __device__ __forceinline__ void split4(const float* f, v4bf* hi, v4bf* lo) {
    #pragma unroll
    for (int j = 0; j < 4; ++j) {
        float v = f[j];
        __bf16 h = (__bf16)v;
        (*hi)[j] = h;
        (*lo)[j] = (__bf16)(v - (float)h);
    }
}

// -------------------------------------------------------------------------
// Kernel 1: aTa[b] = A^T @ A  (A = x[b] as [N=4096, C=256]).
// Block: 8 waves, 64(c) x 128(d) tile; wave = 32x32 (2x2 WMMA subtiles).
// LDS holds a K=32 chunk transposed: row = channel, col = K (n), hi/lo planes.
// Grid: (32 batches, 8): cblk = (y>>1)*64, dblk = (y&1)*128.
// -------------------------------------------------------------------------
__global__ __launch_bounds__(256) void cam_aTa_kernel(
    const float* __restrict__ x, float* __restrict__ aTa)
{
    const int b    = blockIdx.x;
    const int cblk = (blockIdx.y >> 1) * 64;    // A slice: 64 channels
    const int dblk = (blockIdx.y & 1) * 128;    // B slice: 128 channels

    __shared__ __align__(16) __bf16 sHi[192][LDP];   // rows 0..63: c-slice, 64..191: d-slice
    __shared__ __align__(16) __bf16 sLo[192][LDP];

    const int tid  = threadIdx.x;
    const int w    = tid >> 5;
    const int lane = tid & 31;
    const int ml   = lane & 15;
    const int hs   = lane >> 4;

    const int wc = w >> 2;     // 0..1 : c 32-group
    const int wd = w & 3;      // 0..3 : d 32-group
    const int rA0 = wc * 32 + ml;            // sHi rows for A subtiles (i*16 offset)
    const int rB0 = 64 + wd * 32 + ml;       // sHi rows for B subtiles
    const int kA  = hs * 8;                  // A: V0-3=K kA..kA+7, V4-7=+16
    const int kB  = hs * 8;                  // B: V0-3=K kB..kB+7, V4-7=+16

    const float* xb = x + (size_t)b * N_ * C_;

    v8f acc[2][2] = {};

    for (int n0 = 0; n0 < N_; n0 += 32) {
        __syncthreads();
        // Stage 32 K-rows x 192 channels, transposed + hi/lo split.
        // Unit = 4 rows x 4 cols; 384 units over 256 threads.
        #pragma unroll
        for (int i = 0; i < 2; ++i) {
            const int q = tid + i * 256;
            if (q < 384) {
                const int rg  = q / 48;              // 0..7  -> rows rg*4..+3
                const int cg  = q % 48;              // 0..47 -> cols cg*4..+3
                const int col = cg * 4;              // logical channel 0..191
                const int gc  = (col < 64) ? (cblk + col) : (dblk + col - 64);
                float4 qv[4];
                #pragma unroll
                for (int j = 0; j < 4; ++j)
                    qv[j] = *(const float4*)(xb + (size_t)(n0 + rg * 4 + j) * C_ + gc);
                const float* qf = (const float*)qv;  // qf[j*4 + i] = row j, col i
                #pragma unroll
                for (int ci = 0; ci < 4; ++ci) {
                    v4bf ph, pl;
                    float f4[4] = { qf[0*4+ci], qf[1*4+ci], qf[2*4+ci], qf[3*4+ci] };
                    split4(f4, &ph, &pl);
                    *(v4bf*)&sHi[col + ci][rg * 4] = ph;
                    *(v4bf*)&sLo[col + ci][rg * 4] = pl;
                }
            }
        }
        if (n0 + 32 < N_)
            __builtin_prefetch(xb + (size_t)(n0 + 32 + (tid >> 4)) * C_ +
                                   cblk + ((tid & 15) << 2), 0, 1);
        __syncthreads();

        v16bf ahi[2], alo[2], bhi[2], blo[2];
        #pragma unroll
        for (int i = 0; i < 2; ++i) {
            const int r = rA0 + i * 16;
            ahi[i] = ld16_split(&sHi[r][kA], &sHi[r][kA + 16]);
            alo[i] = ld16_split(&sLo[r][kA], &sLo[r][kA + 16]);
        }
        #pragma unroll
        for (int j = 0; j < 2; ++j) {
            const int r = rB0 + j * 16;
            bhi[j] = ld16_split(&sHi[r][kB], &sHi[r][kB + 16]);
            blo[j] = ld16_split(&sLo[r][kB], &sLo[r][kB + 16]);
        }
        #pragma unroll
        for (int i = 0; i < 2; ++i)
            #pragma unroll
            for (int j = 0; j < 2; ++j) {
                acc[i][j] = wmma_bf(ahi[i], bhi[j], acc[i][j]);
                acc[i][j] = wmma_bf(ahi[i], blo[j], acc[i][j]);
                acc[i][j] = wmma_bf(alo[i], bhi[j], acc[i][j]);
            }
    }

    float* outb = aTa + (size_t)b * C_ * C_;
    #pragma unroll
    for (int i = 0; i < 2; ++i)
        #pragma unroll
        for (int j = 0; j < 2; ++j) {
            const int d = dblk + wd * 32 + j * 16 + ml;
            #pragma unroll
            for (int r = 0; r < 8; ++r) {
                const int c = cblk + wc * 32 + i * 16 + r + 8 * hs;
                outb[(size_t)c * C_ + d] = acc[i][j][r];
            }
        }
}

// -------------------------------------------------------------------------
// Kernel 2: in-place row softmax over attn[32*256][256]. One wave per row.
// -------------------------------------------------------------------------
__global__ __launch_bounds__(256) void cam_softmax_kernel(float* __restrict__ attn)
{
    const int row  = blockIdx.x * 8 + (threadIdx.x >> 5);
    const int lane = threadIdx.x & 31;
    float* p = attn + (size_t)row * C_;

    float v[8];
    float m = -INFINITY;
    #pragma unroll
    for (int i = 0; i < 8; ++i) { v[i] = p[lane + 32 * i]; m = fmaxf(m, v[i]); }
    #pragma unroll
    for (int off = 16; off > 0; off >>= 1) m = fmaxf(m, __shfl_xor(m, off, 32));
    float s = 0.0f;
    #pragma unroll
    for (int i = 0; i < 8; ++i) { v[i] = __expf(v[i] - m); s += v[i]; }
    #pragma unroll
    for (int off = 16; off > 0; off >>= 1) s += __shfl_xor(s, off, 32);
    const float inv = 1.0f / s;
    #pragma unroll
    for (int i = 0; i < 8; ++i) p[lane + 32 * i] = v[i] * inv;
}

// -------------------------------------------------------------------------
// Kernel 3: out[b,n,d] = gamma * (A @ attn)[n,d] + x[b,n,d].
// Block: 8 waves, 64(n) x 128(d); wave = 32x32 (2x2 subtiles). K = C = 256
// in chunks of 32. sX: x rows (K contiguous in memory, stored row-major);
// sW: attn staged transposed (row = d, col = K), hi/lo planes.
// Grid: (128, 32): nblk = (x>>1)*64, dblk = (x&1)*128.
// -------------------------------------------------------------------------
__global__ __launch_bounds__(256) void cam_out_kernel(
    const float* __restrict__ x, const float* __restrict__ attn,
    const float* __restrict__ gamma, float* __restrict__ out)
{
    const int b    = blockIdx.y;
    const int nblk = (blockIdx.x >> 1) * 64;
    const int dblk = (blockIdx.x & 1) * 128;

    __shared__ __align__(16) __bf16 sXhi[64][LDP],  sXlo[64][LDP];
    __shared__ __align__(16) __bf16 sWhi[128][LDP], sWlo[128][LDP];

    const int tid  = threadIdx.x;
    const int w    = tid >> 5;
    const int lane = tid & 31;
    const int ml   = lane & 15;
    const int hs   = lane >> 4;

    const int wn = w >> 2;                 // 0..1 : n 32-group
    const int wd = w & 3;                  // 0..3 : d 32-group
    const int rX0 = wn * 32 + ml;
    const int rW0 = wd * 32 + ml;
    const int kA  = hs * 8;
    const int kB  = hs * 8;

    const float* xb = x    + (size_t)b * N_ * C_;
    const float* ab = attn + (size_t)b * C_ * C_;

    v8f acc[2][2] = {};

    for (int kc = 0; kc < C_; kc += 32) {
        __syncthreads();
        // sX: 64 n-rows x 32 k, row-major (K contiguous). 512 float4, 2/thread.
        #pragma unroll
        for (int i = 0; i < 2; ++i) {
            const int q  = tid + i * 256;
            const int r  = q >> 3;               // 0..63
            const int k4 = (q & 7) << 2;         // 0..28
            float4 v = *(const float4*)(xb + (size_t)(nblk + r) * C_ + kc + k4);
            v4bf ph, pl;
            split4((const float*)&v, &ph, &pl);
            *(v4bf*)&sXhi[r][k4] = ph;
            *(v4bf*)&sXlo[r][k4] = pl;
        }
        // sW: 128 d-cols x 32 k, transposed. Unit = 4 k-rows x 4 d-cols, 1/thread.
        {
            const int cg  = tid & 31;            // d col-group
            const int kg  = tid >> 5;            // k row-group (0..7)
            const int col = cg * 4;
            float4 qv[4];
            #pragma unroll
            for (int j = 0; j < 4; ++j)
                qv[j] = *(const float4*)(ab + (size_t)(kc + kg * 4 + j) * C_ + dblk + col);
            const float* qf = (const float*)qv;
            #pragma unroll
            for (int ci = 0; ci < 4; ++ci) {
                v4bf ph, pl;
                float f4[4] = { qf[0*4+ci], qf[1*4+ci], qf[2*4+ci], qf[3*4+ci] };
                split4(f4, &ph, &pl);
                *(v4bf*)&sWhi[col + ci][kg * 4] = ph;
                *(v4bf*)&sWlo[col + ci][kg * 4] = pl;
            }
        }
        __syncthreads();

        v16bf ahi[2], alo[2], bhi[2], blo[2];
        #pragma unroll
        for (int i = 0; i < 2; ++i) {
            const int r = rX0 + i * 16;
            ahi[i] = ld16_split(&sXhi[r][kA], &sXhi[r][kA + 16]);
            alo[i] = ld16_split(&sXlo[r][kA], &sXlo[r][kA + 16]);
        }
        #pragma unroll
        for (int j = 0; j < 2; ++j) {
            const int r = rW0 + j * 16;
            bhi[j] = ld16_split(&sWhi[r][kB], &sWhi[r][kB + 16]);
            blo[j] = ld16_split(&sWlo[r][kB], &sWlo[r][kB + 16]);
        }
        #pragma unroll
        for (int i = 0; i < 2; ++i)
            #pragma unroll
            for (int j = 0; j < 2; ++j) {
                acc[i][j] = wmma_bf(ahi[i], bhi[j], acc[i][j]);
                acc[i][j] = wmma_bf(ahi[i], blo[j], acc[i][j]);
                acc[i][j] = wmma_bf(alo[i], bhi[j], acc[i][j]);
            }
    }

    const float g = gamma[0];
    float* ob = out + (size_t)b * N_ * C_;
    #pragma unroll
    for (int i = 0; i < 2; ++i)
        #pragma unroll
        for (int j = 0; j < 2; ++j) {
            const int d = dblk + wd * 32 + j * 16 + ml;
            #pragma unroll
            for (int r = 0; r < 8; ++r) {
                const int n = nblk + wn * 32 + i * 16 + r + 8 * hs;
                const size_t idx = (size_t)n * C_ + d;
                ob[idx] = g * acc[i][j][r] + xb[idx];
            }
        }
}

// -------------------------------------------------------------------------
extern "C" void kernel_launch(void* const* d_in, const int* in_sizes, int n_in,
                              void* d_out, int out_size, void* d_ws, size_t ws_size,
                              hipStream_t stream)
{
    const float* x     = (const float*)d_in[0];
    const float* gamma = (const float*)d_in[1];
    float* out = (float*)d_out;
    float* aTa = (float*)d_ws;   // 32*256*256 floats = 8 MB scratch

    cam_aTa_kernel    <<<dim3(B_, 8), 256, 0, stream>>>(x, aTa);
    cam_softmax_kernel<<<dim3((B_ * C_) / 8), 256, 0, stream>>>(aTa);
    cam_out_kernel    <<<dim3(64 * 2, B_), 256, 0, stream>>>(x, aTa, gamma, out);
}
